// MoERouter_91147795955939
// MI455X (gfx1250) — matrix operand order repaired and checked by reading, not compiled, not run on hardware
//
#include <hip/hip_runtime.h>
#include <hip/hip_bf16.h>

typedef __attribute__((ext_vector_type(8)))  float   v8f;
typedef __attribute__((ext_vector_type(2)))  __bf16  v2bf;
typedef __attribute__((ext_vector_type(8)))  __bf16  v8bf;
typedef __attribute__((ext_vector_type(16))) __bf16  v16bf;

namespace {
constexpr int kTokens = 16384;   // B*S = 4*4096
constexpr int kH      = 2048;    // hidden dim (GEMM K)
constexpr int kE      = 512;     // experts (GEMM N)
constexpr int kTopK   = 10;
constexpr int BM = 128, BN = 128, BK = 64;
constexpr int kTiles = kH / BK;  // 32
}

// fp32 pair -> packed bf16.
#if __has_builtin(__builtin_amdgcn_cvt_pk_bf16_f32)
__device__ __forceinline__ unsigned pack_bf16(float a, float b) {
    union { v2bf v; unsigned u; } c;
    c.v = __builtin_amdgcn_cvt_pk_bf16_f32(a, b);
    return c.u;
}
#else
// round-to-nearest (+0x8000) then merge high halves with one v_perm_b32:
// result[15:0] = hi16(a), result[31:16] = hi16(b)  -> 3 VALU ops total.
__device__ __forceinline__ unsigned pack_bf16(float a, float b) {
    union { float f; unsigned u; } ua, ub;
    ua.f = a; ub.f = b;
    unsigned ra = ua.u + 0x8000u;
    unsigned rb = ub.u + 0x8000u;
    return __builtin_amdgcn_perm(rb, ra, 0x07060302u);
}
#endif

// ---------------------------------------------------------------------------
// Kernel 1: router logits GEMM  [kTokens,kH] x [kE,kH]^T -> [kTokens,kE]
// bf16 WMMA, fp32 accumulation. 8 waves/block, 128x128 tile/block,
// 64x32 (4x2 accumulators) per wave, BK=64, double-buffered LDS pipeline.
// ---------------------------------------------------------------------------
__global__ __launch_bounds__(256)
void router_gemm_wmma(const float* __restrict__ X,   // [kTokens][kH]
                      const float* __restrict__ W,   // [kE][kH]
                      float* __restrict__ logits)    // [kTokens][kE]
{
    __shared__ __align__(16) unsigned As[2][BM * BK / 2]; // bf16 [BM][BK]
    __shared__ __align__(16) unsigned Bs[2][BN * BK / 2]; // bf16 [BN][BK]

    const int tid  = threadIdx.x;
    const int wave = tid >> 5;
    const int lane = tid & 31;
    const int lr   = lane & 15;     // row/col within 16
    const int lh   = lane >> 4;     // half-wave select

    const int m0 = blockIdx.y * BM;
    const int n0 = blockIdx.x * BN;
    const int wm = (wave >> 2) * 64;   // wave M offset: 0 or 64
    const int wn = (wave &  3) * 32;   // wave N offset: 0,32,64,96

    // staging: thread t owns row = t/2, k-halve (t&1)*32..+31 (8x float4)
    const int srow  = tid >> 1;        // 0..127
    const int shalf = tid & 1;
    const float* aSrc = X + (size_t)(m0 + srow) * kH + shalf * 32;
    const float* bSrc = W + (size_t)(n0 + srow) * kH + shalf * 32;
    const int dstOff  = srow * (BK / 2) + shalf * 16;   // uints

    float4 aReg[8], bReg[8];

    auto issue_loads = [&](int kt) {
        const float4* a4 = (const float4*)(aSrc + kt * BK);
        const float4* b4 = (const float4*)(bSrc + kt * BK);
#pragma unroll
        for (int i = 0; i < 8; ++i) { aReg[i] = a4[i]; bReg[i] = b4[i]; }
    };
    auto store_tile = [&](int buf) {
        unsigned* ad = &As[buf][dstOff];
        unsigned* bd = &Bs[buf][dstOff];
#pragma unroll
        for (int i = 0; i < 8; ++i) {
            ad[i * 2 + 0] = pack_bf16(aReg[i].x, aReg[i].y);
            ad[i * 2 + 1] = pack_bf16(aReg[i].z, aReg[i].w);
            bd[i * 2 + 0] = pack_bf16(bReg[i].x, bReg[i].y);
            bd[i * 2 + 1] = pack_bf16(bReg[i].z, bReg[i].w);
        }
    };

    v8f acc[4][2] = {};

    issue_loads(0);
    store_tile(0);
    __syncthreads();

    for (int kt = 0; kt < kTiles; ++kt) {
        const int cur = kt & 1;
        if (kt + 1 < kTiles) {
            issue_loads(kt + 1);                        // overlap with WMMA below
            __builtin_prefetch(aSrc + (kt + 2) * BK, 0, 3);  // global_prefetch_b8
            __builtin_prefetch(bSrc + (kt + 2) * BK, 0, 3);
        }

        const __bf16* Ab = (const __bf16*)&As[cur][0];
        const __bf16* Bb = (const __bf16*)&Bs[cur][0];

#pragma unroll
        for (int kk = 0; kk < 2; ++kk) {                // two 32-wide K chunks
            // B fragments: lane n<16 holds K=kk*32+0..15 of col N=n,
            // lanes 16-31 hold K=kk*32+16..31 (contiguous: Bs is expert-major)
            v16bf bF[2];
#pragma unroll
            for (int ni = 0; ni < 2; ++ni) {
                const __bf16* rp = Bb + (wn + ni * 16 + lr) * BK + kk * 32;
                v8bf lo = *(const v8bf*)(rp + lh * 16);
                v8bf hi = *(const v8bf*)(rp + lh * 16 + 8);
                bF[ni] = __builtin_shufflevector(lo, hi,
                    0, 1, 2, 3, 4, 5, 6, 7, 8, 9, 10, 11, 12, 13, 14, 15);
            }
            // A fragments: lanes 0-15 hold K=0..15, lanes 16-31
            // hold K=16..31 of row M (contiguous, matches B)
#pragma unroll
            for (int mi = 0; mi < 4; ++mi) {
                const __bf16* rp = Ab + (wm + mi * 16 + lr) * BK + kk * 32;
                v8bf lo = *(const v8bf*)(rp + lh * 16);
                v8bf hi = *(const v8bf*)(rp + lh * 16 + 8);
                v16bf aF = __builtin_shufflevector(lo, hi,
                    0, 1, 2, 3, 4, 5, 6, 7, 8, 9, 10, 11, 12, 13, 14, 15);
                acc[mi][0] = __builtin_amdgcn_wmma_f32_16x16x32_bf16(
                    false, aF, false, bF[0], (short)0, acc[mi][0], false, false);
                acc[mi][1] = __builtin_amdgcn_wmma_f32_16x16x32_bf16(
                    false, aF, false, bF[1], (short)0, acc[mi][1], false, false);
            }
        }

        if (kt + 1 < kTiles) store_tile(cur ^ 1);       // fill other buffer
        __syncthreads();
    }

    // C layout: lanes 0-15: VGPR r -> (M=r, N=lane); lanes 16-31: (M=8+r, N=lane-16)
#pragma unroll
    for (int mi = 0; mi < 4; ++mi)
#pragma unroll
        for (int ni = 0; ni < 2; ++ni)
#pragma unroll
            for (int r = 0; r < 8; ++r) {
                int row = m0 + wm + mi * 16 + lh * 8 + r;
                int col = n0 + wn + ni * 16 + lr;
                logits[(size_t)row * kE + col] = acc[mi][ni][r];
            }
}

// ---------------------------------------------------------------------------
// Kernel 2: per-token top-10 (on biased logits), softmax of unbiased logits,
// and expert-count histogram. One wave32 per token.
// ---------------------------------------------------------------------------
__global__ __launch_bounds__(256)
void router_topk(const float* __restrict__ logits,
                 const float* __restrict__ bias,
                 int*   __restrict__ selOut,   // [kTokens][kTopK]
                 float* __restrict__ wOut,     // [kTokens][kTopK]
                 float* __restrict__ counts)   // [kE]
{
    const int wave  = threadIdx.x >> 5;
    const int lane  = threadIdx.x & 31;
    const int token = blockIdx.x * 8 + wave;
    if (token >= kTokens) return;

    const float* lp = logits + (size_t)token * kE;
    float v[16], b[16];
#pragma unroll
    for (int j = 0; j < 16; ++j) {
        int e = j * 32 + lane;
        v[j] = lp[e];
        b[j] = v[j] + bias[e];
    }

    int   selIdx[kTopK];
    float selVal[kTopK];

#pragma unroll
    for (int k = 0; k < kTopK; ++k) {
        float bestB = -3.4e38f;
        float bestV = 0.f;
        int   bestI = 0x7FFFFFFF;
#pragma unroll
        for (int j = 0; j < 16; ++j) {
            int e = j * 32 + lane;
            if (b[j] > bestB || (b[j] == bestB && e < bestI)) {
                bestB = b[j]; bestV = v[j]; bestI = e;
            }
        }
        // wave32 butterfly argmax, lower index wins ties
#pragma unroll
        for (int off = 16; off > 0; off >>= 1) {
            float oB = __shfl_xor(bestB, off, 32);
            float oV = __shfl_xor(bestV, off, 32);
            int   oI = __shfl_xor(bestI, off, 32);
            if (oB > bestB || (oB == bestB && oI < bestI)) {
                bestB = oB; bestV = oV; bestI = oI;
            }
        }
        selIdx[k] = bestI;
        selVal[k] = bestV;
        if ((bestI & 31) == lane) b[bestI >> 5] = -3.4e38f;  // mask winner
    }

    // fp32 softmax over the K unbiased logits (replicated in all lanes)
    float m = selVal[0];
#pragma unroll
    for (int k = 1; k < kTopK; ++k) m = fmaxf(m, selVal[k]);
    float ex[kTopK];
    float s = 0.f;
#pragma unroll
    for (int k = 0; k < kTopK; ++k) { ex[k] = __expf(selVal[k] - m); s += ex[k]; }
    float inv = 1.0f / s;

    if (lane < kTopK) {
        selOut[token * kTopK + lane] = selIdx[lane];
        wOut[token * kTopK + lane]   = ex[lane] * inv;
        atomicAdd(&counts[selIdx[lane]], 1.0f);   // global_atomic_add_f32
    }
}

// ---------------------------------------------------------------------------
// Kernel 3: loss-free balancing bias update
// ---------------------------------------------------------------------------
__global__ void router_bias_update(const float* __restrict__ bias,
                                   const float* __restrict__ counts,
                                   float* __restrict__ newBias)
{
    int e = blockIdx.x * blockDim.x + threadIdx.x;
    if (e < kE) {
        const float expected = (float)((long long)kTokens * kTopK) / (float)kE; // 320
        float err = expected - counts[e];
        float sgn = (err > 0.f) ? 1.f : ((err < 0.f) ? -1.f : 0.f);
        newBias[e] = bias[e] + 0.001f * sgn;
    }
}

extern "C" void kernel_launch(void* const* d_in, const int* in_sizes, int n_in,
                              void* d_out, int out_size, void* d_ws, size_t ws_size,
                              hipStream_t stream) {
    const float* X    = (const float*)d_in[0];   // hidden_states [4,4096,2048]
    const float* W    = (const float*)d_in[1];   // gate_weight   [512,2048]
    const float* bias = (const float*)d_in[2];   // expert_bias   [512]

    // d_out: logits | selected_experts(int32 bits) | routing_weights | new_bias
    float* out    = (float*)d_out;
    float* logits = out;
    int*   sel    = (int*)(out + (size_t)kTokens * kE);
    float* rw     = out + (size_t)kTokens * kE + (size_t)kTokens * kTopK;
    float* nb     = rw + (size_t)kTokens * kTopK;

    float* counts = (float*)d_ws;                // 512 floats of scratch
    hipMemsetAsync(counts, 0, kE * sizeof(float), stream);

    dim3 gGemm(kE / BN, kTokens / BM);           // (4, 128)
    router_gemm_wmma<<<gGemm, 256, 0, stream>>>(X, W, logits);

    router_topk<<<kTokens / 8, 256, 0, stream>>>(logits, bias, sel, rw, counts);

    router_bias_update<<<2, 256, 0, stream>>>(bias, counts, nb);
}
